// Attention_51307679318359
// MI455X (gfx1250) — compile-verified, matrix operand order checked by reading, not run on hardware
//
#include <hip/hip_runtime.h>
#include <hip/hip_bf16.h>

typedef __attribute__((ext_vector_type(16))) __bf16 v16bf;
typedef __attribute__((ext_vector_type(8)))  float  v8f;
typedef __attribute__((ext_vector_type(4)))  unsigned int u32x4;
typedef __attribute__((ext_vector_type(8)))  unsigned int u32x8;

static constexpr int B  = 2;
static constexpr int S  = 2048;
static constexpr int D  = 2048;
static constexpr int H  = 16;
static constexpr int HD = 128;
static constexpr float NEG_INF = -1.0e9f;
static constexpr float INV_SQRT_HD = 0.08838834764831845f; // 1/sqrt(128)

// ---------- helpers ----------

struct __align__(16) U4 { unsigned int a, b, c, d; };
struct __align__(16) Frag32 { U4 lo, hi; };

__device__ __forceinline__ v16bf make_frag(U4 lo, U4 hi) {
    Frag32 f; f.lo = lo; f.hi = hi;
    return __builtin_bit_cast(v16bf, f);
}

__device__ __forceinline__ unsigned short f2bf(float f) {
    unsigned int u = __builtin_bit_cast(unsigned int, f);
    u += 0x7FFFu + ((u >> 16) & 1u);           // round-to-nearest-even
    return (unsigned short)(u >> 16);
}

// Issue one TDM 2D tile load: 128 rows x 32 bf16 elements, row stride K elems,
// into LDS at byte offset ldsOff with HW padding 16B per 64B row
// (=> LDS row stride 80B = LDT 40 elements). Descriptor per ISA 8.3-8.6.
// `ldsObj` is passed as an asm operand (inside a comment) purely so the LDS
// object escapes -- otherwise alias analysis deletes the tile loads.
__device__ __forceinline__ void tdm_load_tile(const unsigned short* gsrc,
                                              unsigned ldsOff,
                                              unsigned K, unsigned rows,
                                              const void* ldsObj) {
    unsigned long long ga = (unsigned long long)(size_t)gsrc;
    u32x4 g0;
    g0[0] = 1u;                                        // count=1, user mode
    g0[1] = ldsOff;                                    // lds_addr
    g0[2] = (unsigned)(ga & 0xFFFFFFFFu);              // global_addr[31:0]
    g0[3] = (unsigned)((ga >> 32) & 0x01FFFFFFu)       // global_addr[56:32]
          | (2u << 30);                                // type=2 (image)
    u32x8 g1;
    g1[0] = (1u << 16)                                 // data_size=2B
          | (1u << 20)                                 // pad_enable
          | (3u << 22)                                 // pad_interval: 16 DW (64B)
          | (3u << 25);                                // pad_amount: 4 DW (16B)
    g1[1] = (K & 0xFFFFu) << 16;                       // tensor_dim0[15:0]
    g1[2] = (K >> 16) | ((rows & 0xFFFFu) << 16);      // dim0[31:16] | dim1[15:0]
    g1[3] = (rows >> 16) | (32u << 16);                // dim1[31:16] | tile_dim0=32
    g1[4] = 128u;                                      // tile_dim1=128, tile_dim2=0
    g1[5] = K;                                         // tensor_dim0_stride[31:0]
    g1[6] = 0u;                                        // stride0[47:32] | stride1 lo
    g1[7] = 0u;
    u32x4 gz = { 0u, 0u, 0u, 0u };                     // groups 2/3: dims 2+ unused
    asm volatile("tensor_load_to_lds %0, %1, %2, %3 ; lds %4"
                 :: "s"(g0), "s"(g1), "s"(gz), "s"(gz), "s"(ldsObj)
                 : "memory");
}

// ---------- kernel 1: fp32 -> bf16 cast ----------

__global__ void cast_f32_bf16(const float* __restrict__ src,
                              unsigned short* __restrict__ dst, int n) {
    int i = blockIdx.x * blockDim.x + threadIdx.x;
    if (i < n) dst[i] = f2bf(src[i]);
}

// ---------- kernel 2: C[M,N](f32) = A[M,K](bf16) * W[N,K](bf16)^T ----------
// block: 256 threads (8 waves as 2x4), block tile 128x128, wave tile 64x32.
// Tiles staged to LDS by the Tensor Data Mover, double-buffered so the DMA of
// tile k+1 overlaps the WMMA work on tile k (TENSORcnt pipelining).

__launch_bounds__(256)
__global__ void gemm_bf16_tn(const unsigned short* __restrict__ A,
                             const unsigned short* __restrict__ W,
                             float* __restrict__ C,
                             int M, int N, int K) {
    constexpr int LDT   = 40;                   // padded LDS row stride (elems)
    constexpr int TILEB = 128 * LDT;            // elems per tile buffer (10240 B)
    __shared__ __align__(16) unsigned short Tiles[4 * TILEB]; // A0 B0 A1 B1

    const int t    = threadIdx.x;
    const int lane = t & 31;
    const int wave = t >> 5;
    const int wm   = (wave >> 2) * 64;          // 0 or 64
    const int wn   = (wave & 3) * 32;           // 0,32,64,96
    const int m0   = blockIdx.y * 128;
    const int n0   = blockIdx.x * 128;
    const int lr   = lane & 15;
    const int hi8  = (lane < 16) ? 0 : 8;

    // uniform (scalar) wave id so the TDM issue is a truly scalar branch:
    // TDM ignores EXEC, so it must not be vector-predicated.
    const unsigned uwave = (unsigned)__builtin_amdgcn_readfirstlane(wave);

    v8f acc[4][2] = {};

    // prologue: DMA first K-tile pair into buffer 0
    if (uwave == 0) {
        tdm_load_tile(A + (size_t)m0 * K, 0u,     (unsigned)K, (unsigned)M, Tiles);
        tdm_load_tile(W + (size_t)n0 * K, 10240u, (unsigned)K, (unsigned)N, Tiles);
    }

    int cur = 0;
    for (int kt = 0; kt < K; kt += 32) {
        if (uwave == 0) {
            if (kt + 32 < K) {
                // prefetch next tile pair into the other buffer, then wait for
                // the current pair only (2 descriptors still in flight).
                unsigned nb = (unsigned)(cur ^ 1) * 20480u;
                tdm_load_tile(A + (size_t)m0 * K + kt + 32, nb,
                              (unsigned)K, (unsigned)M, Tiles);
                tdm_load_tile(W + (size_t)n0 * K + kt + 32, nb + 10240u,
                              (unsigned)K, (unsigned)N, Tiles);
                __builtin_amdgcn_s_wait_tensorcnt(2);
            } else {
                __builtin_amdgcn_s_wait_tensorcnt(0);
            }
        }
        __syncthreads();                        // current tiles visible to all

        const unsigned short* As = Tiles + cur * (2 * TILEB);
        const unsigned short* Bs = As + TILEB;

        v16bf af[4], bf[2];
        for (int mi = 0; mi < 4; mi++) {
            const unsigned short* p = &As[(wm + mi * 16 + lr) * LDT + hi8];
            af[mi] = make_frag(*(const U4*)p, *(const U4*)(p + 16));
        }
        for (int ni = 0; ni < 2; ni++) {
            const unsigned short* p = &Bs[(wn + ni * 16 + lr) * LDT + hi8];
            bf[ni] = make_frag(*(const U4*)p, *(const U4*)(p + 16));
        }
        for (int mi = 0; mi < 4; mi++)
            for (int ni = 0; ni < 2; ni++)
                acc[mi][ni] = __builtin_amdgcn_wmma_f32_16x16x32_bf16(
                    false, af[mi], false, bf[ni], (short)0, acc[mi][ni],
                    false, false);

        __syncthreads();                        // reads done; buffer reusable
        cur ^= 1;
    }

    for (int mi = 0; mi < 4; mi++)
        for (int ni = 0; ni < 2; ni++) {
            int n = n0 + wn + ni * 16 + lr;
            for (int j = 0; j < 8; j++) {
                int m = m0 + wm + mi * 16 + hi8 + j;
                C[(size_t)m * N + n] = acc[mi][ni][j];
            }
        }
}

// ---------- kernel 3a: RoPE + repack Q,K: [B,S,H,HD]f32 -> [B,H,S,HD]bf16 ----

__global__ void rope_pack_qk(const float* __restrict__ Qraw,
                             const float* __restrict__ Kraw,
                             const float* __restrict__ cosT,
                             const float* __restrict__ sinT,
                             const int*   __restrict__ start_p,
                             unsigned short* __restrict__ Qbf,
                             unsigned short* __restrict__ Kbf) {
    int idx = blockIdx.x * blockDim.x + threadIdx.x;   // B*S*H*(HD/2)
    if (idx >= B * S * H * (HD / 2)) return;
    int i = idx & 63;                 // HD/2 = 64
    int h = (idx >> 6) & 15;          // H = 16
    int s = (idx >> 10) & 2047;       // S = 2048
    int b = idx >> 21;
    int sp = start_p[0];
    float c  = cosT[(sp + s) * (HD / 2) + i];
    float sn = sinT[(sp + s) * (HD / 2) + i];
    size_t src = (size_t)(b * S + s) * D + h * HD + 2 * i;
    float q0 = Qraw[src], q1 = Qraw[src + 1];
    float k0 = Kraw[src], k1 = Kraw[src + 1];
    size_t dst = ((size_t)(b * H + h) * S + s) * HD + 2 * i;
    Qbf[dst]     = f2bf(q0 * c - q1 * sn);
    Qbf[dst + 1] = f2bf(q0 * sn + q1 * c);
    Kbf[dst]     = f2bf(k0 * c - k1 * sn);
    Kbf[dst + 1] = f2bf(k0 * sn + k1 * c);
}

// ---------- kernel 3b: pack V: [B,S,H,HD]f32 -> [B,H,HD,S]bf16 (transposed) --

__global__ void pack_v(const float* __restrict__ Vraw,
                       unsigned short* __restrict__ Vt) {
    int idx = blockIdx.x * blockDim.x + threadIdx.x;   // B*S*D
    if (idx >= B * S * D) return;
    int d = idx & 2047;
    int s = (idx >> 11) & 2047;
    int b = idx >> 22;
    int dd = d & 127;
    int h  = d >> 7;
    Vt[((size_t)(b * H + h) * HD + dd) * S + s] = f2bf(Vraw[idx]);
}

// ---------- kernel 4: causal flash attention, bf16 WMMA ----------
// block: 128 threads (4 waves); each wave owns 16 query rows x full HD=128.

__launch_bounds__(128)
__global__ void flash_attn(const unsigned short* __restrict__ Qbf,
                           const unsigned short* __restrict__ Kbf,
                           const unsigned short* __restrict__ Vt,
                           unsigned short* __restrict__ Obf) {
    constexpr int LP = 40;
    __shared__ __align__(16) unsigned short Pl[4][16 * LP]; // per-wave P scratch

    const int lane = threadIdx.x & 31;
    const int wave = threadIdx.x >> 5;
    const int lr   = lane & 15;
    const int hi8  = (lane < 16) ? 0 : 8;
    const int b    = blockIdx.z;
    const int h    = blockIdx.y;
    const int q0   = blockIdx.x * 64 + wave * 16;
    const size_t bh = (size_t)(b * H + h);
    const unsigned short* Qh = Qbf + bh * S * HD;
    const unsigned short* Kh = Kbf + bh * S * HD;
    const unsigned short* Vh = Vt + bh * HD * S;

    // resident Q fragments: 16 rows x HD, 4 K-steps of 32
    v16bf qf[4];
    for (int kk = 0; kk < 4; kk++) {
        const unsigned short* p = Qh + (size_t)(q0 + lr) * HD + kk * 32 + hi8;
        qf[kk] = make_frag(*(const U4*)p, *(const U4*)(p + 16));
    }

    v8f accO[8] = {};
    float mrow[8], lsum[8];
    for (int j = 0; j < 8; j++) { mrow[j] = -1.0e30f; lsum[j] = 0.0f; }

    unsigned short* Pw = &Pl[wave][0];

    for (int kt = 0; kt < q0 + 16; kt += 32) {
        // prefetch next key/value tiles (global_prefetch_b8)
        if (kt + 32 < q0 + 16) {
            __builtin_prefetch(Kh + (size_t)(kt + 32 + lr) * HD, 0, 3);
            __builtin_prefetch(Vh + (size_t)(lane * 4) * S + kt + 32, 0, 3);
        }

        // ---- S = Q * K^T for 32 keys (two 16-key C tiles) ----
        v8f s0 = {}, s1 = {};
        for (int kk = 0; kk < 4; kk++) {
            const unsigned short* p0 = Kh + (size_t)(kt + lr) * HD + kk * 32 + hi8;
            v16bf k0f = make_frag(*(const U4*)p0, *(const U4*)(p0 + 16));
            s0 = __builtin_amdgcn_wmma_f32_16x16x32_bf16(
                false, qf[kk], false, k0f, (short)0, s0, false, false);
            const unsigned short* p1 = Kh + (size_t)(kt + 16 + lr) * HD + kk * 32 + hi8;
            v16bf k1f = make_frag(*(const U4*)p1, *(const U4*)(p1 + 16));
            s1 = __builtin_amdgcn_wmma_f32_16x16x32_bf16(
                false, qf[kk], false, k1f, (short)0, s1, false, false);
        }

        // ---- scale + causal mask + online softmax ----
        const int kn0 = kt + lr;
        const int kn1 = kt + 16 + lr;
        float scale[8];
        for (int j = 0; j < 8; j++) {
            int qm = q0 + hi8 + j;
            float v0 = s0[j] * INV_SQRT_HD + ((kn0 <= qm) ? 0.0f : NEG_INF);
            float v1 = s1[j] * INV_SQRT_HD + ((kn1 <= qm) ? 0.0f : NEG_INF);
            float tmax = fmaxf(v0, v1);
            for (int off = 1; off < 16; off <<= 1)
                tmax = fmaxf(tmax, __shfl_xor(tmax, off, 32));
            float mn = fmaxf(mrow[j], tmax);
            float p0 = __expf(v0 - mn);
            float p1 = __expf(v1 - mn);
            float rs = p0 + p1;
            for (int off = 1; off < 16; off <<= 1)
                rs += __shfl_xor(rs, off, 32);
            scale[j] = __expf(mrow[j] - mn);
            lsum[j]  = lsum[j] * scale[j] + rs;
            mrow[j]  = mn;
            // stash P (C-layout -> LDS) for A-fragment reload
            Pw[(hi8 + j) * LP + lr]      = f2bf(p0);
            Pw[(hi8 + j) * LP + 16 + lr] = f2bf(p1);
        }
        for (int o = 0; o < 8; o++)
            for (int j = 0; j < 8; j++) accO[o][j] *= scale[j];

        asm volatile("s_wait_dscnt 0" ::: "memory");   // wave-private LDS RAW

        const unsigned short* pp = Pw + lr * LP + hi8;
        v16bf pf = make_frag(*(const U4*)pp, *(const U4*)(pp + 16));

        // ---- O += P * V (V^T rows contiguous) ----
        for (int o = 0; o < 8; o++) {
            const unsigned short* vp = Vh + (size_t)(o * 16 + lr) * S + kt + hi8;
            v16bf vf = make_frag(*(const U4*)vp, *(const U4*)(vp + 16));
            accO[o] = __builtin_amdgcn_wmma_f32_16x16x32_bf16(
                false, pf, false, vf, (short)0, accO[o], false, false);
        }
    }

    // ---- normalize + store bf16 to [B,S,D] for the output projection ----
    for (int o = 0; o < 8; o++) {
        int n = h * HD + o * 16 + lr;
        for (int j = 0; j < 8; j++) {
            int m = q0 + hi8 + j;
            Obf[(size_t)(b * S + m) * D + n] = f2bf(accO[o][j] / lsum[j]);
        }
    }
}

// ---------- host ----------

extern "C" void kernel_launch(void* const* d_in, const int* in_sizes, int n_in,
                              void* d_out, int out_size, void* d_ws, size_t ws_size,
                              hipStream_t stream) {
    (void)in_sizes; (void)n_in; (void)out_size; (void)ws_size;

    const float* x    = (const float*)d_in[0];
    const float* wq   = (const float*)d_in[1];
    const float* wk   = (const float*)d_in[2];
    const float* wv   = (const float*)d_in[3];
    const float* wo   = (const float*)d_in[4];
    const float* cosT = (const float*)d_in[5];
    const float* sinT = (const float*)d_in[6];
    // d_in[7] = mask (causal tril; computed analytically in-kernel)
    const int*   sp   = (const int*)d_in[8];
    float* out = (float*)d_out;

    // workspace layout (bytes, all 256-aligned)
    constexpr size_t SZ_XBF  = (size_t)B * S * D * 2;   // 16 MiB
    constexpr size_t SZ_W    = (size_t)D * D * 2;       //  8 MiB each
    constexpr size_t SZ_RAW  = (size_t)B * S * D * 4;   // 32 MiB each
    constexpr size_t SZ_PBF  = (size_t)B * S * D * 2;   // 16 MiB each
    constexpr size_t OFF_XBF  = 0;
    constexpr size_t OFF_WQ   = OFF_XBF  + SZ_XBF;
    constexpr size_t OFF_WK   = OFF_WQ   + SZ_W;
    constexpr size_t OFF_WV   = OFF_WK   + SZ_W;
    constexpr size_t OFF_WO   = OFF_WV   + SZ_W;
    constexpr size_t OFF_QRAW = OFF_WO   + SZ_W;
    constexpr size_t OFF_KRAW = OFF_QRAW + SZ_RAW;
    constexpr size_t OFF_VRAW = OFF_KRAW + SZ_RAW;
    constexpr size_t OFF_QBF  = OFF_VRAW + SZ_RAW;
    constexpr size_t OFF_KBF  = OFF_QBF  + SZ_PBF;
    constexpr size_t OFF_VT   = OFF_KBF  + SZ_PBF;
    constexpr size_t OFF_ATTN = OFF_VT   + SZ_PBF;

    char* ws = (char*)d_ws;
    unsigned short* Xbf  = (unsigned short*)(ws + OFF_XBF);
    unsigned short* WQb  = (unsigned short*)(ws + OFF_WQ);
    unsigned short* WKb  = (unsigned short*)(ws + OFF_WK);
    unsigned short* WVb  = (unsigned short*)(ws + OFF_WV);
    unsigned short* WOb  = (unsigned short*)(ws + OFF_WO);
    float*          Qraw = (float*)(ws + OFF_QRAW);
    float*          Kraw = (float*)(ws + OFF_KRAW);
    float*          Vraw = (float*)(ws + OFF_VRAW);
    unsigned short* Qbf  = (unsigned short*)(ws + OFF_QBF);
    unsigned short* Kbf  = (unsigned short*)(ws + OFF_KBF);
    unsigned short* Vt   = (unsigned short*)(ws + OFF_VT);
    unsigned short* Attn = (unsigned short*)(ws + OFF_ATTN);

    // 1) casts
    const int nx = B * S * D;
    const int nw = D * D;
    cast_f32_bf16<<<(nx + 255) / 256, 256, 0, stream>>>(x,  Xbf, nx);
    cast_f32_bf16<<<(nw + 255) / 256, 256, 0, stream>>>(wq, WQb, nw);
    cast_f32_bf16<<<(nw + 255) / 256, 256, 0, stream>>>(wk, WKb, nw);
    cast_f32_bf16<<<(nw + 255) / 256, 256, 0, stream>>>(wv, WVb, nw);
    cast_f32_bf16<<<(nw + 255) / 256, 256, 0, stream>>>(wo, WOb, nw);

    // 2) QKV projections (WMMA GEMM, double-buffered TDM tiles)
    dim3 gg(D / 128, (B * S) / 128);
    gemm_bf16_tn<<<gg, 256, 0, stream>>>(Xbf, WQb, Qraw, B * S, D, D);
    gemm_bf16_tn<<<gg, 256, 0, stream>>>(Xbf, WKb, Kraw, B * S, D, D);
    gemm_bf16_tn<<<gg, 256, 0, stream>>>(Xbf, WVb, Vraw, B * S, D, D);

    // 3) RoPE + layout packing
    const int np = B * S * H * (HD / 2);
    rope_pack_qk<<<(np + 255) / 256, 256, 0, stream>>>(Qraw, Kraw, cosT, sinT,
                                                       sp, Qbf, Kbf);
    pack_v<<<(nx + 255) / 256, 256, 0, stream>>>(Vraw, Vt);

    // 4) causal flash attention
    dim3 ga(S / 64, H, B);
    flash_attn<<<ga, 128, 0, stream>>>(Qbf, Kbf, Vt, Attn);

    // 5) output projection -> f32 d_out
    gemm_bf16_tn<<<gg, 256, 0, stream>>>(Attn, WOb, out, B * S, D, D);
}
